// AttentionAggregator_33698313404804
// MI455X (gfx1250) — matrix operand order repaired and checked by reading, not compiled, not run on hardware
//
#include <hip/hip_runtime.h>

#define BATCH   20000
#define KNBR    10
#define FDIM    256
#define HDIM    128
#define NODES   8                 // batch nodes per block
#define ROWS    (NODES * KNBR)    // 80 rows = 5 exact 16-row M tiles
#define MTILES  (ROWS / 16)       // 5
#define NTILES  (HDIM / 16)       // 8
#define THREADS 256               // 8 waves (wave32) -> 8 N-tiles

#define KT1     (FDIM / 32)       // 8 wmma K-steps, GEMM1
#define KT2     (HDIM / 32)       // 4 wmma K-steps, GEMM2

// pre-swizzled weight fragment sizes (bf16 elements)
#define WA_FRAG_ELEMS (NTILES * KT1 * 32 * 16)   // 32768 per Wa matrix
#define WB_FRAG_ELEMS (NTILES * KT2 * 32 * 16)   // 16384 per Wb matrix
#define WS_ELEMS      (3 * WA_FRAG_ELEMS + 3 * WB_FRAG_ELEMS)  // 147456
#define WS_BYTES      (WS_ELEMS * 2)                           // 294912

typedef __bf16 bf16_t;
typedef bf16_t v16bf __attribute__((ext_vector_type(16)));
typedef float  v8f   __attribute__((ext_vector_type(8)));

union FragAB {
    v16bf  v;
    bf16_t h[16];
    uint4  q[2];
};

#if __has_builtin(__builtin_amdgcn_tanhf)
#define FAST_TANH(x) __builtin_amdgcn_tanhf(x)
#else
#define FAST_TANH(x) tanhf(x)
#endif

// ---------------------------------------------------------------------------
// prep kernel: convert the 6 f32 weight matrices to bf16, laid out in per-lane
// WMMA B-fragment order:  frag f = (ntile*KT + kt), element = ((f*32)+lane)*16+j
// lane<16 -> K = kt*32 + j ; lane>=16 -> K = kt*32 + 16 + j ; col = ntile*16+(lane&15)
// ---------------------------------------------------------------------------
__global__ __launch_bounds__(256, 1)
void prep_weights_kernel(const float* __restrict__ W1a, const float* __restrict__ W1b,
                         const float* __restrict__ W2a, const float* __restrict__ W2b,
                         const float* __restrict__ W3a, const float* __restrict__ W3b,
                         bf16_t* __restrict__ wf)
{
    const int idx = blockIdx.x * 256 + threadIdx.x;      // one (mat, frag, lane)
    const int NA  = 3 * NTILES * KT1 * 32;               // 6144 Wa triples
    const int NB  = 3 * NTILES * KT2 * 32;               // 3072 Wb triples
    if (idx >= NA + NB) return;

    const float* W;
    bf16_t*      dst;
    int t, KT;
    if (idx < NA) {
        const int mat = idx / (NTILES * KT1 * 32);
        t  = idx % (NTILES * KT1 * 32);
        KT = KT1;
        W   = (mat == 0) ? W1a : (mat == 1) ? W2a : W3a;
        dst = wf + (size_t)mat * WA_FRAG_ELEMS;
    } else {
        const int r   = idx - NA;
        const int mat = r / (NTILES * KT2 * 32);
        t  = r % (NTILES * KT2 * 32);
        KT = KT2;
        W   = (mat == 0) ? W1b : (mat == 1) ? W2b : W3b;
        dst = wf + (size_t)(3 * WA_FRAG_ELEMS) + (size_t)mat * WB_FRAG_ELEMS;
    }
    const int lane = t & 31;
    const int f    = t >> 5;
    const int kt   = f % KT;
    const int n    = f / KT;
    const int k0   = kt * 32 + ((lane >> 4) << 4);
    const int col  = n * 16 + (lane & 15);

    FragAB frag;
    #pragma unroll
    for (int j = 0; j < 16; ++j)
        frag.h[j] = (bf16_t)W[(size_t)(k0 + j) * HDIM + col];

    uint4* o = (uint4*)(dst + (size_t)t * 16);
    o[0] = frag.q[0];
    o[1] = frag.q[1];
}

// ---------------------------------------------------------------------------
// fused main kernel
// ---------------------------------------------------------------------------
template<bool PREPPED>
__global__ __launch_bounds__(THREADS, 1)
void attn_agg_kernel(const int* __restrict__ neighbors,
                     const float* __restrict__ embed,
                     const float* __restrict__ W1a, const float* __restrict__ W1b,
                     const float* __restrict__ W2a, const float* __restrict__ W2b,
                     const float* __restrict__ W3a, const float* __restrict__ W3b,
                     const bf16_t* __restrict__ wf,
                     float* __restrict__ out)
{
    __shared__ bf16_t xs[ROWS][FDIM];        // 40 KB: gathered x (bf16)
    __shared__ bf16_t hs[ROWS][HDIM];        // 20 KB: tanh(x@Wa)
    __shared__ bf16_t qkv[3][ROWS][HDIM];    // 60 KB: q,k,v
    __shared__ int   rowsrc[ROWS];
    __shared__ float satt[NODES][KNBR][KNBR];
    __shared__ float swsum[NODES][KNBR];

    const int tid   = threadIdx.x;
    const int wave  = tid >> 5;
    const int lane  = tid & 31;
    const int node0 = blockIdx.x * NODES;

    // ---- stage 0: resolve gather indices ----
    if (tid < ROWS) {
        rowsrc[tid] = neighbors[(node0 + tid / KNBR) * KNBR + (tid % KNBR)];
    }
    __syncthreads();

    // ---- stage 1: gather embeddings, f32 -> bf16 into LDS ----
    for (int i = tid; i < ROWS * (FDIM / 4); i += THREADS) {
        const int r  = i >> 6;               // FDIM/4 == 64
        const int c4 = (i & 63) << 2;
        const float4 f = *(const float4*)(embed + (size_t)rowsrc[r] * FDIM + c4);
        union { bf16_t h[4]; uint2 u; } pk;
        pk.h[0] = (bf16_t)f.x; pk.h[1] = (bf16_t)f.y;
        pk.h[2] = (bf16_t)f.z; pk.h[3] = (bf16_t)f.w;
        *(uint2*)&xs[r][c4] = pk.u;
    }

    const int half = lane >> 4;
    const int l16  = lane & 15;
    const int ncol = wave * 16 + l16;

    const float*  Wa[3]  = { W1a, W2a, W3a };
    const float*  Wb[3]  = { W1b, W2b, W3b };

    for (int br = 0; br < 3; ++br) {
        __syncthreads();  // xs ready / hs free for reuse

        // ===== GEMM1: h = tanh(x @ Wa) =====
        FragAB bfr[KT1];
        if (PREPPED) {
            const bf16_t* sec = wf + (size_t)br * WA_FRAG_ELEMS
                              + ((size_t)(wave * KT1) * 32 + lane) * 16;
            #pragma unroll
            for (int kt = 0; kt < KT1; ++kt) {
                const uint4* p = (const uint4*)(sec + (size_t)kt * 32 * 16);
                bfr[kt].q[0] = p[0];
                bfr[kt].q[1] = p[1];
            }
        } else {
            #pragma unroll
            for (int kt = 0; kt < KT1; ++kt) {
                const float* wp = Wa[br] + (size_t)(kt * 32 + half * 16) * HDIM + ncol;
                #pragma unroll
                for (int j = 0; j < 16; ++j)
                    bfr[kt].h[j] = (bf16_t)wp[j * HDIM];
            }
        }
        #pragma unroll
        for (int m = 0; m < MTILES; ++m) {
            v8f acc = {};
            const int arow = m * 16 + l16;
            #pragma unroll
            for (int kt = 0; kt < KT1; ++kt) {
                FragAB a;
                const bf16_t* ap = &xs[arow][kt * 32 + half * 16];
                a.q[0] = *(const uint4*)(ap);
                a.q[1] = *(const uint4*)(ap + 8);
                acc = __builtin_amdgcn_wmma_f32_16x16x32_bf16(
                        false, a.v, false, bfr[kt].v, (short)0, acc, false, false);
            }
            #pragma unroll
            for (int g = 0; g < 8; ++g) {
                const int row = m * 16 + g + half * 8;
                hs[row][ncol] = (bf16_t)FAST_TANH(acc[g]);
            }
        }
        __syncthreads();  // hs ready

        // ===== GEMM2: qkv[br] = h @ Wb =====
        FragAB bfr2[KT2];
        if (PREPPED) {
            const bf16_t* sec = wf + (size_t)(3 * WA_FRAG_ELEMS)
                              + (size_t)br * WB_FRAG_ELEMS
                              + ((size_t)(wave * KT2) * 32 + lane) * 16;
            #pragma unroll
            for (int kt = 0; kt < KT2; ++kt) {
                const uint4* p = (const uint4*)(sec + (size_t)kt * 32 * 16);
                bfr2[kt].q[0] = p[0];
                bfr2[kt].q[1] = p[1];
            }
        } else {
            #pragma unroll
            for (int kt = 0; kt < KT2; ++kt) {
                const float* wp = Wb[br] + (size_t)(kt * 32 + half * 16) * HDIM + ncol;
                #pragma unroll
                for (int j = 0; j < 16; ++j)
                    bfr2[kt].h[j] = (bf16_t)wp[j * HDIM];
            }
        }
        #pragma unroll
        for (int m = 0; m < MTILES; ++m) {
            v8f acc = {};
            const int arow = m * 16 + l16;
            #pragma unroll
            for (int kt = 0; kt < KT2; ++kt) {
                FragAB a;
                const bf16_t* ap = &hs[arow][kt * 32 + half * 16];
                a.q[0] = *(const uint4*)(ap);
                a.q[1] = *(const uint4*)(ap + 8);
                acc = __builtin_amdgcn_wmma_f32_16x16x32_bf16(
                        false, a.v, false, bfr2[kt].v, (short)0, acc, false, false);
            }
            #pragma unroll
            for (int g = 0; g < 8; ++g) {
                const int row = m * 16 + g + half * 8;
                qkv[br][row][ncol] = (bf16_t)acc[g];
            }
        }
    }
    __syncthreads();  // q,k,v complete

    // ===== attention epilogue: one wave per node =====
    const int r0 = wave * KNBR;

    for (int p = lane; p < KNBR * KNBR; p += 32) {
        const int i = p / KNBR, j = p % KNBR;
        const bf16_t* qi = qkv[0][r0 + i];
        const bf16_t* kj = qkv[1][r0 + j];
        float s = 0.f;
        #pragma unroll 8
        for (int d = 0; d < HDIM; ++d)
            s += (float)qi[d] * (float)kj[d];
        satt[wave][i][j] = s;
    }
    __syncthreads();

    if (lane < KNBR) {
        const int i = lane;
        float mx = -3.4e38f;
        #pragma unroll
        for (int j = 0; j < KNBR; ++j) mx = fmaxf(mx, satt[wave][i][j]);
        float e[KNBR], sum = 0.f;
        #pragma unroll
        for (int j = 0; j < KNBR; ++j) { e[j] = __expf(satt[wave][i][j] - mx); sum += e[j]; }
        const float inv = 1.0f / sum;
        #pragma unroll
        for (int j = 0; j < KNBR; ++j) satt[wave][i][j] = e[j] * inv;
    }
    __syncthreads();

    if (lane < KNBR) {
        const int j = lane;
        float s = 0.f;
        #pragma unroll
        for (int i = 0; i < KNBR; ++i) s += satt[wave][i][j];
        swsum[wave][j] = s;
    }
    __syncthreads();

    for (int h = lane; h < HDIM; h += 32) {
        float o = 0.f;
        #pragma unroll
        for (int j = 0; j < KNBR; ++j)
            o += swsum[wave][j] * (float)qkv[2][r0 + j][h];
        out[(size_t)(node0 + wave) * HDIM + h] = o;
    }
}

extern "C" void kernel_launch(void* const* d_in, const int* in_sizes, int n_in,
                              void* d_out, int out_size, void* d_ws, size_t ws_size,
                              hipStream_t stream) {
    (void)in_sizes; (void)n_in; (void)out_size;
    const int*   neighbors = (const int*)d_in[0];
    const float* embed     = (const float*)d_in[1];
    const float* W1a       = (const float*)d_in[2];
    const float* W1b       = (const float*)d_in[3];
    const float* W2a       = (const float*)d_in[4];
    const float* W2b       = (const float*)d_in[5];
    const float* W3a       = (const float*)d_in[6];
    const float* W3b       = (const float*)d_in[7];
    float*       out       = (float*)d_out;

    const dim3 grid(BATCH / NODES);   // 2500 blocks

    if (d_ws != nullptr && ws_size >= (size_t)WS_BYTES) {
        bf16_t* wf = (bf16_t*)d_ws;
        const int prep_threads = 3 * NTILES * KT1 * 32 + 3 * NTILES * KT2 * 32; // 9216
        prep_weights_kernel<<<(prep_threads + 255) / 256, 256, 0, stream>>>(
            W1a, W1b, W2a, W2b, W3a, W3b, wf);
        attn_agg_kernel<true><<<grid, THREADS, 0, stream>>>(
            neighbors, embed, W1a, W1b, W2a, W2b, W3a, W3b, wf, out);
    } else {
        attn_agg_kernel<false><<<grid, THREADS, 0, stream>>>(
            neighbors, embed, W1a, W1b, W2a, W2b, W3a, W3b, nullptr, out);
    }
}
